// SparseLookupFFNv3_20547123544589
// MI455X (gfx1250) — matrix operand order repaired and checked by reading, not compiled, not run on hardware
//
#include <hip/hip_runtime.h>
#include <hip/hip_bf16.h>
#include <math.h>

// ---------------------------------------------------------------------------
// SparseLookupFFNv3 for gfx1250 (MI455X).
//   - All four GEMMs run through v_wmma_f32_16x16x32_bf16 (fp32 accum).
//   - GEMM: 128Mx64N block tile, 8 waves, 32x32 per-wave register tile (2x2
//     WMMA accumulators), K-step 32, A/B staged in LDS with double buffering.
//   - LDS staging uses GLOBAL_LOAD_ASYNC_TO_LDS_B128 (ASYNCcnt-tracked).
// ---------------------------------------------------------------------------

#define NROWS (8 * 4096)
#define DD    1024
#define HCC   256
#define TLEN  4096

#define BLK_M 128
#define BLK_N 64
#define BLK_K 32

#define GLOBAL_AS __attribute__((address_space(1)))
#define LDS_AS    __attribute__((address_space(3)))

#if defined(__has_builtin)
#  if __has_builtin(__builtin_amdgcn_global_load_async_to_lds_b128)
#    define USE_ASYNC_COPY 1
#  endif
#endif
#ifndef USE_ASYNC_COPY
#  define USE_ASYNC_COPY 0
#endif

#if defined(__has_builtin) && __has_builtin(__builtin_amdgcn_s_wait_asynccnt)
#  define WAIT_ASYNC() __builtin_amdgcn_s_wait_asynccnt(0)
#else
#  define WAIT_ASYNC() asm volatile("s_wait_asynccnt 0" ::: "memory")
#endif

typedef __attribute__((ext_vector_type(16))) __bf16        v16bf;
typedef __attribute__((ext_vector_type(8)))  float         v8f;
typedef __attribute__((ext_vector_type(8)))  unsigned int  v8u;
typedef int b128_t __attribute__((vector_size(16)));   // async b128 transfer type

static __device__ __forceinline__ unsigned short f2bf(float f) {
  unsigned int u = __float_as_uint(f);
  u = (u + 0x7FFFu + ((u >> 16) & 1u)) >> 16;   // round-to-nearest-even
  return (unsigned short)u;
}

static __device__ __forceinline__ float signf(float x) {
  return (x > 0.f) ? 1.f : ((x < 0.f) ? -1.f : 0.f);
}

static __device__ __forceinline__ float bspline(float t) {
  t = fabsf(t);
  if (t < 1.f) return 2.f / 3.f - t * t + 0.5f * t * t * t;
  if (t < 2.f) { float u = 2.f - t; return u * u * u * (1.f / 6.f); }
  return 0.f;
}

static __device__ __forceinline__ float gelu_exact(float x) {
  return 0.5f * x * (1.f + erff(x * 0.70710678118654752f));
}

static __device__ __forceinline__ float quant_coeff(float c) {
  return (c > 0.3f) ? 1.f : ((c < -0.3f) ? -1.f : 0.f);
}

// ---------------------------------------------------------------------------
// Prep kernels
// ---------------------------------------------------------------------------

// ST has 128 rows (padded): 0..63 sign(dir), 64..71 csig, 72..127 zero.
__global__ __launch_bounds__(256) void k_build_S(const float* __restrict__ dir,
                                                 unsigned short* __restrict__ ST) {
  int idx = blockIdx.x * 256 + threadIdx.x;
  if (idx >= 128 * DD) return;
  int t = idx / DD, k = idx % DD;
  float v;
  if (t < 64) {
    v = signf(dir[(size_t)t * DD + k]);
  } else if (t < 72) {
    int c = t - 64;
    float s = 0.f;
    for (int i = 0; i < 8; ++i) s += signf(dir[(size_t)(c * 8 + i) * DD + k]);
    v = signf(s);
  } else {
    v = 0.f;
  }
  ST[idx] = f2bf(v);
}

// dst[n*K + k] = bf16(src[k*Nc + n])  (B^T, row-major Nc x K)
__global__ __launch_bounds__(256) void k_transpose_bf16(const float* __restrict__ src,
                                                        unsigned short* __restrict__ dst,
                                                        int K, int Nc) {
  int idx = blockIdx.x * 256 + threadIdx.x;
  if (idx >= K * Nc) return;
  int n = idx / K, k = idx % K;
  dst[idx] = f2bf(src[(size_t)k * Nc + n]);
}

// ---------------------------------------------------------------------------
// LayerNorm: one wave per row
// ---------------------------------------------------------------------------
__global__ __launch_bounds__(256) void k_layernorm(const float* __restrict__ x,
                                                   const float* __restrict__ g,
                                                   const float* __restrict__ b,
                                                   float* __restrict__ xnF,
                                                   unsigned short* __restrict__ xnH) {
  int wid = threadIdx.x >> 5, lane = threadIdx.x & 31;
  int row = blockIdx.x * 8 + wid;
  if (row >= NROWS) return;
  const float* xr = x + (size_t)row * DD;
  float s = 0.f, s2 = 0.f;
  for (int d = lane; d < DD; d += 32) { float v = xr[d]; s += v; s2 += v * v; }
  for (int o = 16; o; o >>= 1) { s += __shfl_xor(s, o, 32); s2 += __shfl_xor(s2, o, 32); }
  float mu = s * (1.f / DD);
  float var = s2 * (1.f / DD) - mu * mu;
  float rstd = rsqrtf(var + 1e-5f);
  for (int d = lane; d < DD; d += 32) {
    float v = (xr[d] - mu) * rstd * g[d] + b[d];
    xnF[(size_t)row * DD + d] = v;
    xnH[(size_t)row * DD + d] = f2bf(v);
  }
}

// ---------------------------------------------------------------------------
// LDS fragment loads (layouts per CDNA5 ISA §7.12.2)
// ---------------------------------------------------------------------------
static __device__ __forceinline__ v16bf frag_a_lds(const unsigned short* __restrict__ base,
                                                   int hl) {
  // base = &s_a[row_local * BLK_K]; A 16-bit layout: k = 2(v&3) + 8*hl + 16*(v>>2)
  v8u u;
#pragma unroll
  for (int v = 0; v < 8; ++v) {
    int kk = ((v & 3) << 1) + (hl << 3) + ((v >> 2) << 4);
    u[v] = *reinterpret_cast<const unsigned int*>(base + kk);
  }
  return __builtin_bit_cast(v16bf, u);
}

static __device__ __forceinline__ v16bf frag_b_lds(const unsigned short* __restrict__ base) {
  // base = &s_b[col_local * BLK_K + 8*hl]; B: k = 2(r&3) + 8*hl + 16*(r>>2)
  v8u u;
#pragma unroll
  for (int r = 0; r < 8; ++r) {
    u[r] = *reinterpret_cast<const unsigned int*>(base + ((r & 3) << 1) + ((r >> 2) << 4));
  }
  return __builtin_bit_cast(v16bf, u);
}

// ---------------------------------------------------------------------------
// WMMA bf16 GEMM: out[M x Nc] = A[M x K] @ BT[Nc x K]^T
// mode: 0 = f32 (acc+bias), 1 = gelu->f32, 2 = gelu->bf16
// ---------------------------------------------------------------------------
__global__ __launch_bounds__(256) void k_wmma_gemm_bf16(const unsigned short* __restrict__ A,
                                                        const unsigned short* __restrict__ BT,
                                                        float* __restrict__ outF,
                                                        unsigned short* __restrict__ outH,
                                                        const float* __restrict__ bias,
                                                        int Nc, int K, int mode) {
  __shared__ unsigned short s_a[2][BLK_M * BLK_K];   // 2 x 8 KB
  __shared__ unsigned short s_b[2][BLK_N * BLK_K];   // 2 x 4 KB

  const int tid  = threadIdx.x;
  const int lane = tid & 31;
  const int wid  = tid >> 5;
  const int hl   = lane >> 4;
  const int m16  = lane & 15;
  const int wm   = wid & 3;      // 4 waves along M (32 rows each)
  const int wn   = wid >> 2;     // 2 waves along N (32 cols each)
  const int rowBase = blockIdx.y * BLK_M;
  const int colBase = blockIdx.x * BLK_N;

  // cooperative staging indices: A = 128x32 (16 bf16/thread), B = 64x32 (8/thread)
  const int sa_row = tid >> 1;            // 0..127
  const int sa_k   = (tid & 1) << 4;      // 0 / 16
  const int sb_col = tid >> 2;            // 0..63
  const int sb_k   = (tid & 3) << 3;      // 0/8/16/24

  const unsigned short* gA = A  + (size_t)(rowBase + sa_row) * K + sa_k;
  const unsigned short* gB = BT + (size_t)(colBase + sb_col) * K + sb_k;

  auto stage = [&](int buf, int k0) {
#if USE_ASYNC_COPY
    __builtin_amdgcn_global_load_async_to_lds_b128(
        (GLOBAL_AS b128_t*)(gA + k0),
        (LDS_AS b128_t*)&s_a[buf][sa_row * BLK_K + sa_k], 0, 0);
    __builtin_amdgcn_global_load_async_to_lds_b128(
        (GLOBAL_AS b128_t*)(gA + k0),
        (LDS_AS b128_t*)&s_a[buf][sa_row * BLK_K + sa_k], 16, 0);
    __builtin_amdgcn_global_load_async_to_lds_b128(
        (GLOBAL_AS b128_t*)(gB + k0),
        (LDS_AS b128_t*)&s_b[buf][sb_col * BLK_K + sb_k], 0, 0);
#else
    *(uint4*)&s_a[buf][sa_row * BLK_K + sa_k]     = *(const uint4*)(gA + k0);
    *(uint4*)&s_a[buf][sa_row * BLK_K + sa_k + 8] = *(const uint4*)(gA + k0 + 8);
    *(uint4*)&s_b[buf][sb_col * BLK_K + sb_k]     = *(const uint4*)(gB + k0);
#endif
  };

  v8f acc[2][2];
#pragma unroll
  for (int mt = 0; mt < 2; ++mt)
#pragma unroll
    for (int nt = 0; nt < 2; ++nt)
      acc[mt][nt] = v8f{0.f, 0.f, 0.f, 0.f, 0.f, 0.f, 0.f, 0.f};

  stage(0, 0);
  int cur = 0;
  for (int k0 = 0; k0 < K; k0 += BLK_K) {
#if USE_ASYNC_COPY
    WAIT_ASYNC();
#endif
    __syncthreads();
    if (k0 + BLK_K < K) stage(cur ^ 1, k0 + BLK_K);

    v16bf afrag[2], bfrag[2];
#pragma unroll
    for (int mt = 0; mt < 2; ++mt) {
      int rloc = wm * 32 + mt * 16 + m16;
      afrag[mt] = frag_a_lds(&s_a[cur][rloc * BLK_K], hl);
    }
#pragma unroll
    for (int nt = 0; nt < 2; ++nt) {
      int cloc = wn * 32 + nt * 16 + m16;
      bfrag[nt] = frag_b_lds(&s_b[cur][cloc * BLK_K + (hl << 3)]);
    }
#pragma unroll
    for (int mt = 0; mt < 2; ++mt)
#pragma unroll
      for (int nt = 0; nt < 2; ++nt)
        acc[mt][nt] = __builtin_amdgcn_wmma_f32_16x16x32_bf16(
            false, afrag[mt], false, bfrag[nt], (short)0, acc[mt][nt], false, false);

    cur ^= 1;
  }

#pragma unroll
  for (int mt = 0; mt < 2; ++mt) {
#pragma unroll
    for (int nt = 0; nt < 2; ++nt) {
      int col = colBase + wn * 32 + nt * 16 + m16;
      if (col < Nc) {
        float bv = bias ? bias[col] : 0.f;
#pragma unroll
        for (int r = 0; r < 8; ++r) {
          int row = rowBase + wm * 32 + mt * 16 + r + hl * 8;
          float v = acc[mt][nt][r] + bv;
          size_t o = (size_t)row * Nc + col;
          if (mode == 0)      outF[o] = v;
          else if (mode == 1) outF[o] = gelu_exact(v);
          else                outH[o] = f2bf(gelu_exact(v));
        }
      }
    }
  }
}

// ---------------------------------------------------------------------------
// Tile / cluster selection + gauge application (one wave per row)
// ---------------------------------------------------------------------------
__global__ __launch_bounds__(256) void k_select(const float* __restrict__ scores,
                                                const int* __restrict__ positions,
                                                const float* __restrict__ xnF,
                                                const float* __restrict__ gscale,
                                                const float* __restrict__ gshift,
                                                unsigned short* __restrict__ xlH,
                                                int* __restrict__ tidxA,
                                                float* __restrict__ twA) {
  int wid = threadIdx.x >> 5, lane = threadIdx.x & 31;
  int row = blockIdx.x * 8 + wid;
  if (row >= NROWS) return;

  int tsel = 0;
  if (lane == 0) {
    float p = (float)positions[row % TLEN];
    // cluster argmax  (w_c = SPREAD*TPC/NT = 0.25)
    int cidx = 0; float best = -INFINITY;
    for (int j = 0; j < 8; ++j) {
      float cp = bspline((p * (8.f / 4096.f) - (float)j) * 4.f);
      float v = scores[(size_t)row * 96 + 64 + j] * cp;
      if (v > best) { best = v; cidx = j; }
    }
    // tile argmax within cluster (tpos spread = 2.0)
    int tbest = cidx * 8; float bw = -INFINITY;
    for (int t = cidx * 8; t < cidx * 8 + 8; ++t) {
      float tp = bspline((p * (64.f / 4096.f) - (float)t) * 0.5f);
      float comb = scores[(size_t)row * 96 + t] * tp;
      if (comb > bw) { bw = comb; tbest = t; }
    }
    tsel = tbest;
    tidxA[row] = tbest;
    twA[row] = bw;
  }
  tsel = __shfl(tsel, 0, 32);

  const float* gs = gscale + (size_t)tsel * DD;
  const float* gb = gshift + (size_t)tsel * DD;
  for (int d = lane; d < DD; d += 32) {
    float v = xnF[(size_t)row * DD + d] * gs[d] + gb[d];
    xlH[(size_t)row * DD + d] = f2bf(v);
  }
}

// ---------------------------------------------------------------------------
// Spline eval + residual (one wave per row)
// ---------------------------------------------------------------------------
__global__ __launch_bounds__(256) void k_spline(const float* __restrict__ h,
                                                const float* __restrict__ Wc2,
                                                const float* __restrict__ bc2,
                                                const float* __restrict__ coeffs,
                                                const float* __restrict__ sscales,
                                                const int* __restrict__ tidxA,
                                                const float* __restrict__ twA,
                                                const float* __restrict__ xnF,
                                                const float* __restrict__ dir,
                                                const float* __restrict__ gscale,
                                                const float* __restrict__ gshift,
                                                unsigned short* __restrict__ resH,
                                                float* __restrict__ spA) {
  int wid = threadIdx.x >> 5, lane = threadIdx.x & 31;
  int row = blockIdx.x * 8 + wid;
  if (row >= NROWS) return;

  const float* hr = h + (size_t)row * HCC;
  float a0 = 0.f, a1 = 0.f;
  for (int j = lane; j < HCC; j += 32) {
    float v = hr[j];
    a0 += v * Wc2[j * 2 + 0];
    a1 += v * Wc2[j * 2 + 1];
  }
  for (int o = 16; o; o >>= 1) { a0 += __shfl_xor(a0, o, 32); a1 += __shfl_xor(a1, o, 32); }

  int t = tidxA[row];
  float sp = 0.f;
  if (lane == 0) {
    float a = tanhf(a0 + bc2[0]);
    float b = tanhf(a1 + bc2[1]);
    int ia = (int)((a + 1.f) * 0.5f * 16.f); ia = ia < 0 ? 0 : (ia > 15 ? 15 : ia);
    int ib = (int)((b + 1.f) * 0.5f * 16.f); ib = ib < 0 ? 0 : (ib > 15 ? 15 : ib);
    const float* c = coeffs + (((size_t)t * 16 + ia) * 16 + ib) * 3;
    float c0 = quant_coeff(c[0]), c1 = quant_coeff(c[1]), c2 = quant_coeff(c[2]);
    const float cs = 2.f / 16.f;
    float la = (a + 1.f - (float)ia * cs) / cs;
    float lb = (b + 1.f - (float)ib * cs) / cs;
    sp = (c0 + c1 * la + c2 * lb) * sscales[t] * twA[row];
    spA[row] = sp;
  }
  sp = __shfl(sp, 0, 32);

  const float* dr = dir + (size_t)t * DD;
  const float* gs = gscale + (size_t)t * DD;
  const float* gb = gshift + (size_t)t * DD;
  for (int d = lane; d < DD; d += 32) {
    float o = sp * dr[d];
    o = (o - gb[d]) / (gs[d] + 1e-6f);
    float r = xnF[(size_t)row * DD + d] - o;
    resH[(size_t)row * DD + d] = f2bf(r);
  }
}

// ---------------------------------------------------------------------------
// Final elementwise combine
// ---------------------------------------------------------------------------
__global__ __launch_bounds__(256) void k_final(const float* __restrict__ x,
                                               const float* __restrict__ ffn,
                                               const float* __restrict__ dir,
                                               const float* __restrict__ gscale,
                                               const float* __restrict__ gshift,
                                               const int* __restrict__ tidxA,
                                               const float* __restrict__ spA,
                                               const float* __restrict__ vgate,
                                               const float* __restrict__ oscale,
                                               float* __restrict__ y) {
  size_t i = (size_t)blockIdx.x * 256 + threadIdx.x;
  if (i >= (size_t)NROWS * DD) return;
  int row = (int)(i >> 10);
  int d = (int)(i & (DD - 1));
  int t = tidxA[row];
  float o = spA[row] * dir[(size_t)t * DD + d];
  o = (o - gshift[(size_t)t * DD + d]) / (gscale[(size_t)t * DD + d] + 1e-6f);
  float gate = 1.f / (1.f + expf(-vgate[0]));
  y[i] = x[i] + (o + ffn[i] * gate) * oscale[0];
}

// ---------------------------------------------------------------------------
// Host launch
// ---------------------------------------------------------------------------
extern "C" void kernel_launch(void* const* d_in, const int* in_sizes, int n_in,
                              void* d_out, int out_size, void* d_ws, size_t ws_size,
                              hipStream_t stream) {
  const float* x        = (const float*)d_in[0];
  const int*   positions= (const int*)  d_in[1];
  const float* ln_g     = (const float*)d_in[2];
  const float* ln_b     = (const float*)d_in[3];
  const float* Wc1      = (const float*)d_in[4];
  const float* bc1      = (const float*)d_in[5];
  const float* Wc2      = (const float*)d_in[6];
  const float* bc2      = (const float*)d_in[7];
  const float* dir      = (const float*)d_in[8];
  const float* coeffs   = (const float*)d_in[9];
  const float* sscales  = (const float*)d_in[10];
  const float* gscale   = (const float*)d_in[11];
  const float* gshift   = (const float*)d_in[12];
  const float* Wv1      = (const float*)d_in[13];
  const float* bv1      = (const float*)d_in[14];
  const float* Wv2      = (const float*)d_in[15];
  const float* bv2      = (const float*)d_in[16];
  const float* vgate    = (const float*)d_in[17];
  const float* oscale   = (const float*)d_in[18];
  float* y = (float*)d_out;

  char* ws = (char*)d_ws;
  size_t off = 0;
  auto take = [&](size_t bytes) -> void* {
    void* p = ws + off;
    off += (bytes + 255) & ~(size_t)255;
    return p;
  };

  const size_t N = NROWS;
  float*          xnF   = (float*)         take(N * DD * 4);   // reused as ffn out
  unsigned short* bfBuf = (unsigned short*)take(N * DD * 2);   // xn_bf16 -> xl -> res
  float*          scores= (float*)         take(N * 96 * 4);
  float*          hF    = (float*)         take(N * HCC * 4);
  unsigned short* hvH   = (unsigned short*)take(N * HCC * 2);
  int*            tidxA = (int*)           take(N * 4);
  float*          twA   = (float*)         take(N * 4);
  float*          spA   = (float*)         take(N * 4);
  unsigned short* ST    = (unsigned short*)take((size_t)128 * DD * 2);
  unsigned short* WcT   = (unsigned short*)take((size_t)HCC * DD * 2);
  unsigned short* WvT   = (unsigned short*)take((size_t)HCC * DD * 2);
  unsigned short* Wv2T  = (unsigned short*)take((size_t)DD * HCC * 2);

  // --- prep: bf16 sign-matrix (padded to 128 rows) + transposed weights ---
  k_build_S<<<(128 * DD + 255) / 256, 256, 0, stream>>>(dir, ST);
  k_transpose_bf16<<<(DD * HCC + 255) / 256, 256, 0, stream>>>(Wc1, WcT, DD, HCC);
  k_transpose_bf16<<<(DD * HCC + 255) / 256, 256, 0, stream>>>(Wv1, WvT, DD, HCC);
  k_transpose_bf16<<<(HCC * DD + 255) / 256, 256, 0, stream>>>(Wv2, Wv2T, HCC, DD);

  // --- layernorm ---
  k_layernorm<<<N / 8, 256, 0, stream>>>(x, ln_g, ln_b, xnF, bfBuf);

  // --- scores = xn @ [sig|csig].T   (N x 96, padded B to 128 rows) ---
  k_wmma_gemm_bf16<<<dim3(2, N / BLK_M), 256, 0, stream>>>(
      bfBuf, ST, scores, nullptr, nullptr, 96, DD, 0);

  // --- select tile, apply gauge (xl overwrites xn_bf16) ---
  k_select<<<N / 8, 256, 0, stream>>>(scores, positions, xnF, gscale, gshift,
                                      bfBuf, tidxA, twA);

  // --- h = gelu(xl @ Wc1 + bc1)   (N x 256, f32) ---
  k_wmma_gemm_bf16<<<dim3(HCC / BLK_N, N / BLK_M), 256, 0, stream>>>(
      bfBuf, WcT, hF, nullptr, bc1, HCC, DD, 1);

  // --- spline + residual (res overwrites xl) ---
  k_spline<<<N / 8, 256, 0, stream>>>(hF, Wc2, bc2, coeffs, sscales, tidxA, twA,
                                      xnF, dir, gscale, gshift, bfBuf, spA);

  // --- hv = gelu(res @ Wv1 + bv1)   (N x 256, bf16) ---
  k_wmma_gemm_bf16<<<dim3(HCC / BLK_N, N / BLK_M), 256, 0, stream>>>(
      bfBuf, WvT, nullptr, hvH, bv1, HCC, DD, 2);

  // --- ffn = hv @ Wv2 + bv2   (N x 1024, f32; reuses xnF buffer) ---
  k_wmma_gemm_bf16<<<dim3(DD / BLK_N, N / BLK_M), 256, 0, stream>>>(
      hvH, Wv2T, xnF, nullptr, bv2, DD, HCC, 0);

  // --- final combine ---
  k_final<<<(unsigned)((N * DD + 255) / 256), 256, 0, stream>>>(
      x, xnF, dir, gscale, gshift, tidxA, spA, vgate, oscale, y);
}
